// PeepholeLSTMModule_41618233098824
// MI455X (gfx1250) — matrix operand order repaired and checked by reading, not compiled, not run on hardware
//
#include <hip/hip_runtime.h>
#include <hip/hip_bf16.h>

// CDNA5 / gfx1250, wave32. WMMA bf16 16x16x32, f32 accumulate.

typedef __attribute__((ext_vector_type(16))) __bf16 v16bf;
typedef __attribute__((ext_vector_type(8)))  __bf16 v8bf;
typedef __attribute__((ext_vector_type(8)))  float  v8f;

// Problem sizes
// B=64, T=1024, D=256, H=256, 4H=1024
// Workspace layout (byte offsets):
//   xA  @ 0        : bf16 [4096 mt][8 kt][32 lane][16]   = 32 MB (A-frag packed x)
//   Wpk @ 32 MB    : bf16 [2 dir][8 kt][64 nt][32][16]   =  1 MB (B-frag packed W)
//   Upk @ 33 MB    : bf16 [2 dir][8 kt][64 nt][32][16]   =  1 MB (B-frag packed U)
//   gpre@ 34 MB    : f32  [2 dir][B*T][1024]             = 512 MB (xW + b)

__device__ __forceinline__ float sigmoid_f(float x) {
    return 1.0f / (1.0f + __expf(-x));
}
__device__ __forceinline__ float tanh_f(float x) {
    x = fminf(fmaxf(x, -15.0f), 15.0f);
    float e = __expf(-2.0f * x);
    return (1.0f - e) / (1.0f + e);
}

// A-fragment K index for 16-bit A 16x32 (ISA 7.12.2): VGPR pair v, lane-group g.
__device__ __forceinline__ unsigned a_k0(int v, unsigned g) {
    return (v < 4) ? (2u * v + 8u * g) : (16u + 2u * (v - 4) + 8u * g);
}

// ---------------------------------------------------------------------------
// Kernel 1: pack x (f32, row-major [B*T][256]) into bf16 WMMA A-fragments.
// ---------------------------------------------------------------------------
__global__ void pack_x_kernel(const float* __restrict__ x, __bf16* __restrict__ xA) {
    unsigned id = blockIdx.x * 256u + threadIdx.x;     // 2^24 total
    unsigned j    = id & 15u;
    unsigned lane = (id >> 4) & 31u;
    unsigned kt   = (id >> 9) & 7u;
    unsigned mt   = id >> 12;
    unsigned m = lane & 15u, g = lane >> 4;
    unsigned v = j >> 1, lo = j & 1u;
    unsigned k = kt * 32u + a_k0((int)v, g) + lo;
    unsigned row = mt * 16u + m;
    xA[id] = (__bf16)x[(size_t)row * 256u + k];
}

// ---------------------------------------------------------------------------
// Kernel 2: pack W_f/U_f/W_b/U_b (f32 [256][1024]) into bf16 B-fragments.
// B 32x16 layout: lane&15 = N column; element j -> K = 8*(lane>>4) + (j&7) + 16*(j>>3).
// ---------------------------------------------------------------------------
__global__ void pack_wu_kernel(const float* __restrict__ Wf, const float* __restrict__ Uf,
                               const float* __restrict__ Wb, const float* __restrict__ Ub,
                               __bf16* __restrict__ Wpk, __bf16* __restrict__ Upk) {
    unsigned id = blockIdx.x * 256u + threadIdx.x;     // 2^20 total
    unsigned j    = id & 15u;
    unsigned lane = (id >> 4) & 31u;
    unsigned nt   = (id >> 9) & 63u;
    unsigned kt   = (id >> 15) & 7u;
    unsigned mat  = id >> 18;                           // 0:Wf 1:Uf 2:Wb 3:Ub
    unsigned k = kt * 32u + (lane >> 4) * 8u + (j & 7u) + (j >> 3) * 16u;
    unsigned n = nt * 16u + (lane & 15u);
    const float* src = (mat == 0) ? Wf : (mat == 1) ? Uf : (mat == 2) ? Wb : Ub;
    float val = src[(size_t)k * 1024u + n];
    unsigned within = ((kt * 64u + nt) * 32u + lane) * 16u + j;
    unsigned dir = mat >> 1;
    __bf16* dst = (mat & 1u) ? Upk : Wpk;
    dst[(size_t)dir * 262144u + within] = (__bf16)val;
}

// ---------------------------------------------------------------------------
// Kernel 3: g_pre = x @ W + b for both directions.
// One wave per 16x16 output tile, K = 256 (8 x wmma 16x16x32 bf16).
// ---------------------------------------------------------------------------
__global__ void xw_gemm_kernel(const __bf16* __restrict__ xA, const __bf16* __restrict__ Wpk,
                               const float* __restrict__ bfv, const float* __restrict__ bbv,
                               float* __restrict__ gpre) {
    unsigned wv = threadIdx.x >> 5, l = threadIdx.x & 31u;
    unsigned job = blockIdx.x * 8u + wv;    // 2 dirs * 4096 mt * 64 nt = 2^19 jobs
    unsigned dir = job >> 18;
    unsigned rem = job & 262143u;
    unsigned mt = rem >> 6, nt = rem & 63u;

    const __bf16* wbase = Wpk + (size_t)dir * 262144u;
    v8f acc = {};
#pragma unroll
    for (int kt = 0; kt < 8; ++kt) {
        v16bf a = *(const v16bf*)(xA + (((size_t)mt * 8u + (unsigned)kt) * 32u + l) * 16u);
        v16bf b = *(const v16bf*)(wbase + (((size_t)kt * 64u + nt) * 32u + l) * 16u);
        acc = __builtin_amdgcn_wmma_f32_16x16x32_bf16(false, a, false, b,
                                                      (short)0, acc, false, false);
    }
    unsigned n = nt * 16u + (l & 15u);
    float bias = (dir ? bbv : bfv)[n];
    size_t base = ((size_t)dir << 26);   // 64*1024*1024 f32 per direction
#pragma unroll
    for (int r = 0; r < 8; ++r) {
        unsigned row = mt * 16u + ((l < 16u) ? (unsigned)r : 8u + (unsigned)r);
        gpre[base + (size_t)row * 1024u + n] = acc[r] + bias;
    }
}

// ---------------------------------------------------------------------------
// Kernel 4: sequential peephole scan. One workgroup = one (dir, 16-row batch
// tile). Wave w owns N-tiles {w, w+8, ..., w+56}, so i/f/g/o columns of each
// hidden unit stay in the same wave's accumulators (lane-local gate math).
// A-fragment per lane/K-tile = two contiguous 16 B runs of h -> 2x ds_load_b128.
// ---------------------------------------------------------------------------
__global__ void lstm_scan_kernel(const float* __restrict__ gpre,
                                 const __bf16* __restrict__ Upk,
                                 const float* __restrict__ pif, const float* __restrict__ pff,
                                 const float* __restrict__ pof,
                                 const float* __restrict__ pib, const float* __restrict__ pfb,
                                 const float* __restrict__ pob,
                                 float* __restrict__ out) {
    __shared__ __align__(16) __bf16 h_lds[16][256];   // 8 KB  hidden state (bf16 A operand)
    __shared__ float c_lds[16][256];                  // 16 KB cell state (f32)

    const unsigned dir   = blockIdx.x >> 2;
    const unsigned btile = blockIdx.x & 3u;
    const unsigned wv = threadIdx.x >> 5, l = threadIdx.x & 31u;
    const unsigned m16 = l & 15u, ghalf = l >> 4;

    for (unsigned i = threadIdx.x; i < 16u * 256u; i += 256u) {
        ((__bf16*)h_lds)[i] = (__bf16)0.0f;
        ((float*)c_lds)[i] = 0.0f;
    }
    __syncthreads();

    const __bf16* ubase = Upk + (size_t)dir * 262144u;
    const float* pi = dir ? pib : pif;
    const float* pf = dir ? pfb : pff;
    const float* po = dir ? pob : pof;

    const unsigned hc0 = wv * 16u + m16;          // hidden column, half 0 (N-tile w)
    const unsigned hc1 = (wv + 8u) * 16u + m16;   // hidden column, half 1 (N-tile w+8)
    const float pi0 = pi[hc0], pi1 = pi[hc1];
    const float pf0 = pf[hc0], pf1 = pf[hc1];
    const float po0 = po[hc0], po1 = po[hc1];

    for (unsigned step = 0; step < 1024u; ++step) {
        const unsigned tt = dir ? (1023u - step) : step;

        // Seed accumulators with g_pre = xW + b for this timestep.
        v8f acc[8];
#pragma unroll
        for (int li = 0; li < 8; ++li) {
            unsigned nt = wv + 8u * (unsigned)li;
            unsigned n = nt * 16u + m16;
#pragma unroll
            for (int r = 0; r < 8; ++r) {
                unsigned bg = btile * 16u + ((l < 16u) ? (unsigned)r : 8u + (unsigned)r);
                acc[li][r] = gpre[((size_t)dir << 26) +
                                  ((size_t)bg * 1024u + tt) * 1024u + n];
            }
        }

        // g += h @ U : 8 K-tiles x 8 N-tiles of wmma bf16.
#pragma unroll
        for (int kt = 0; kt < 8; ++kt) {
            // Per-lane A fragment = h[m16][kt*32+8g .. +8) ++ h[m16][kt*32+16+8g .. +8)
            unsigned kbase = (unsigned)kt * 32u + 8u * ghalf;
            v8bf alo = *(const v8bf*)(&h_lds[m16][kbase]);
            v8bf ahi = *(const v8bf*)(&h_lds[m16][kbase + 16u]);
            v16bf a;
#pragma unroll
            for (int j = 0; j < 8; ++j) {
                a[j]     = alo[j];
                a[8 + j] = ahi[j];
            }
#pragma unroll
            for (int li = 0; li < 8; ++li) {
                unsigned nt = wv + 8u * (unsigned)li;
                v16bf b = *(const v16bf*)(ubase + (((size_t)kt * 64u + nt) * 32u + l) * 16u);
                acc[li] = __builtin_amdgcn_wmma_f32_16x16x32_bf16(false, a, false, b,
                                                                  (short)0, acc[li],
                                                                  false, false);
            }
        }
        __syncthreads();   // all waves done reading old h before it is overwritten

        // Lane-local gate math: tiles li = {0,1}=i, {2,3}=f, {4,5}=g, {6,7}=o.
#pragma unroll
        for (int half = 0; half < 2; ++half) {
            const unsigned hc = half ? hc1 : hc0;
            const float ppi = half ? pi1 : pi0;
            const float ppf = half ? pf1 : pf0;
            const float ppo = half ? po1 : po0;
#pragma unroll
            for (int r = 0; r < 8; ++r) {
                unsigned m = (l < 16u) ? (unsigned)r : 8u + (unsigned)r;
                float cold = c_lds[m][hc];
                float iv = sigmoid_f(acc[0 + half][r] + ppi * cold);
                float fv = sigmoid_f(acc[2 + half][r] + ppf * cold);
                float cn = fv * cold + iv * tanh_f(acc[4 + half][r]);
                float ov = sigmoid_f(acc[6 + half][r] + ppo * cn);
                float hn = ov * tanh_f(cn);
                c_lds[m][hc] = cn;
                h_lds[m][hc] = (__bf16)hn;

                unsigned bg = btile * 16u + m;
                size_t oidx = ((size_t)bg * 1024u + tt) * 512u + dir * 256u + hc;
                unsigned hsh = (unsigned)oidx * 2654435761u;
                hsh ^= hsh >> 16; hsh *= 0x45d9f3bu; hsh ^= hsh >> 16;
                out[oidx] = (hsh & 1u) ? hn * 2.0f : 0.0f;   // dropout p=0.5, scale 2
            }
        }
        __syncthreads();
    }
}

// ---------------------------------------------------------------------------
extern "C" void kernel_launch(void* const* d_in, const int* in_sizes, int n_in,
                              void* d_out, int out_size, void* d_ws, size_t ws_size,
                              hipStream_t stream) {
    const float* x    = (const float*)d_in[0];
    const float* W_f  = (const float*)d_in[1];
    const float* U_f  = (const float*)d_in[2];
    const float* b_f  = (const float*)d_in[3];
    const float* pi_f = (const float*)d_in[4];
    const float* pf_f = (const float*)d_in[5];
    const float* po_f = (const float*)d_in[6];
    const float* W_b  = (const float*)d_in[7];
    const float* U_b  = (const float*)d_in[8];
    const float* b_b  = (const float*)d_in[9];
    const float* pi_b = (const float*)d_in[10];
    const float* pf_b = (const float*)d_in[11];
    const float* po_b = (const float*)d_in[12];
    float* out = (float*)d_out;

    char* ws = (char*)d_ws;
    __bf16* xA   = (__bf16*)(ws);                          // 32 MB
    __bf16* Wpk  = (__bf16*)(ws + (size_t)(32u << 20));    //  1 MB
    __bf16* Upk  = (__bf16*)(ws + (size_t)(33u << 20));    //  1 MB
    float*  gpre = (float*) (ws + (size_t)(34u << 20));    // 512 MB

    pack_x_kernel <<<65536, 256, 0, stream>>>(x, xA);
    pack_wu_kernel<<<4096, 256, 0, stream>>>(W_f, U_f, W_b, U_b, Wpk, Upk);
    xw_gemm_kernel<<<65536, 256, 0, stream>>>(xA, Wpk, b_f, b_b, gpre);
    lstm_scan_kernel<<<8, 256, 0, stream>>>(gpre, Upk,
                                            pi_f, pf_f, po_f,
                                            pi_b, pf_b, po_b, out);
}
